// GRU_AE_35407710388454
// MI455X (gfx1250) — compile-verified, matrix operand order checked by reading, not run on hardware
//
#include <hip/hip_runtime.h>

// ---------------------------------------------------------------------------
// GRU autoencoder (4 stacked GRUs + FC) for MI455X (gfx1250, wave32, WMMA).
//
//  - Activations staged as f16 [B, T, 32] (stride-32; padded weight columns
//    are zero so padding never contributes).
//  - Per layer: one persistent kernel, 32 blocks x 1 wave, each wave owns a
//    16-row batch tile and scans all T=2048 steps.
//  - Weights repacked so r/z gate rows occupy tiles [0, RZT) and n gate rows
//    occupy tiles [RZT, RZT+NTT). r/z tiles chain the x-GEMM result into the
//    h-GEMM's C operand (D = h@Whh^T + x@Wih^T) -> single LDS store set.
//    n tiles keep gx/gh separate (needed for r * hn).
//  - v_wmma_f32_16x16x32_f16 everywhere (K=32 zero-padded), gates in f32
//    VALU, h kept in registers + mirrored to LDS in WMMA A-layout.
// ---------------------------------------------------------------------------

typedef __attribute__((ext_vector_type(16))) _Float16 v16h;
typedef __attribute__((ext_vector_type(8)))  _Float16 v8h;
typedef __attribute__((ext_vector_type(8)))  float    v8f;

#define B_  512
#define T_  2048
#define NC_ 10
#define FCD 16

__device__ __forceinline__ v16h join8(v8h lo, v8h hi) {
  return __builtin_shufflevector(lo, hi, 0,1,2,3,4,5,6,7,8,9,10,11,12,13,14,15);
}

__device__ __forceinline__ float sigmoidf_(float x) {
  return 1.0f / (1.0f + __expf(-x));
}

// ---------------------------------------------------------------------------
// Repack fp32 weight [3H x K] into zero-padded f16 [(RZP+NP) x 32]:
//   rows [0, 2H)   (r,z gates) -> packed rows [0, RZP)
//   rows [2H, 3H)  (n gate)    -> packed rows [RZP, RZP+NP)
// ---------------------------------------------------------------------------
__global__ void pack_weights_rzn(const float* __restrict__ w, _Float16* __restrict__ wp,
                                 int H, int K, int RZP, int NP) {
  int idx = blockIdx.x * blockDim.x + threadIdx.x;
  if (idx >= (RZP + NP) * 32) return;
  int g = idx >> 5, k = idx & 31;
  float v = 0.0f;
  if (k < K) {
    if (g < RZP) {
      if (g < 2 * H) v = w[g * K + k];
    } else {
      int j = g - RZP;
      if (j < H) v = w[(2 * H + j) * K + k];
    }
  }
  wp[idx] = (_Float16)v;
}

__global__ void f32_to_f16(const float* __restrict__ src, _Float16* __restrict__ dst, int n) {
  int i = blockIdx.x * blockDim.x + threadIdx.x;
  if (i < n) dst[i] = (_Float16)src[i];
}

// ---------------------------------------------------------------------------
// GRU scan layer. HID = hidden size, RZT = ceil(2*HID/16), NTT = ceil(HID/16).
// Block = 1 wave (32 threads), owns batch rows [16*blockIdx.x, +16).
// ---------------------------------------------------------------------------
template <int HID, int RZT, int NTT>
__global__ __launch_bounds__(32) void gru_scan(
    const _Float16* __restrict__ act_in,   // [B,T,32] f16
    _Float16*       __restrict__ act_out,  // [B,T,32] f16 (cols < HID written)
    const _Float16* __restrict__ wih_p,    // [(RZT+NTT)*16][32] f16 packed
    const _Float16* __restrict__ whh_p,    // [(RZT+NTT)*16][32] f16 packed
    const float*    __restrict__ bih,      // [3*HID]
    const float*    __restrict__ bhh)      // [3*HID]
{
  __shared__ float    s_grz[16][RZT * 16];   // gx+gh for r,z gates
  __shared__ float    s_gnx[16][NTT * 16];   // gx for n gate
  __shared__ float    s_gnh[16][NTT * 16];   // gh for n gate
  __shared__ _Float16 s_h[16][32];
  __shared__ float    s_bsum[RZT * 16];      // b_ih + b_hh (r,z)
  __shared__ float    s_bnx[NTT * 16];       // b_ih (n)
  __shared__ float    s_bnh[NTT * 16];       // b_hh (n)

  const int lane  = threadIdx.x;        // 0..31
  const int b0    = blockIdx.x * 16;    // batch tile base
  const int n_loc = lane & 15;
  const int kbase = (lane >> 4) * 8;    // A/B layout: lanes 16-31 start at K=8

  // --- Preload both weight matrices into WMMA B-layout registers ---
  v16h wih_rz[RZT], whh_rz[RZT], wih_n[NTT], whh_n[NTT];
#pragma unroll
  for (int nt = 0; nt < RZT; ++nt) {
    int g = nt * 16 + n_loc;            // B column n == packed weight row g
    wih_rz[nt] = join8(*(const v8h*)(wih_p + g * 32 + kbase),
                       *(const v8h*)(wih_p + g * 32 + kbase + 16));
    whh_rz[nt] = join8(*(const v8h*)(whh_p + g * 32 + kbase),
                       *(const v8h*)(whh_p + g * 32 + kbase + 16));
  }
#pragma unroll
  for (int nt = 0; nt < NTT; ++nt) {
    int g = (RZT + nt) * 16 + n_loc;
    wih_n[nt] = join8(*(const v8h*)(wih_p + g * 32 + kbase),
                      *(const v8h*)(wih_p + g * 32 + kbase + 16));
    whh_n[nt] = join8(*(const v8h*)(whh_p + g * 32 + kbase),
                      *(const v8h*)(whh_p + g * 32 + kbase + 16));
  }

  // --- Zero h state in LDS (padded cols stay zero forever), stage biases ---
  for (int i = lane; i < 16 * 32; i += 32) (&s_h[0][0])[i] = (_Float16)0;
  for (int i = lane; i < RZT * 16; i += 32)
    s_bsum[i] = (i < 2 * HID) ? (bih[i] + bhh[i]) : 0.0f;
  for (int i = lane; i < NTT * 16; i += 32) {
    s_bnx[i] = (i < HID) ? bih[2 * HID + i] : 0.0f;
    s_bnh[i] = (i < HID) ? bhh[2 * HID + i] : 0.0f;
  }
  __syncthreads();

  // Persistent per-thread h registers (fixed (b,j) mapping per loop slot).
  constexpr int GITER = (16 * HID + 31) / 32;
  float hreg[GITER];
#pragma unroll
  for (int i = 0; i < GITER; ++i) hreg[i] = 0.0f;

  const int       brow    = n_loc;      // A row = batch within tile
  const long long rowbase = (long long)(b0 + brow) * T_;

  for (int t = 0; t < T_; ++t) {
    // Gather x A-tile: two contiguous 16B chunks -> global_load_b128 x2.
    const _Float16* xrow = act_in + (rowbase + t) * 32;
    v16h ax = join8(*(const v8h*)(xrow + kbase),
                    *(const v8h*)(xrow + kbase + 16));
    if (t + 1 < T_)
      __builtin_prefetch(act_in + (rowbase + t + 1) * 32, 0, 3);

    // Gather h A-tile from LDS: ds_load_b128 x2.
    const _Float16* hrow = &s_h[brow][0];
    v16h ah = join8(*(const v8h*)(hrow + kbase),
                    *(const v8h*)(hrow + kbase + 16));

    // r/z tiles: D = h @ Whh^T + (x @ Wih^T), single LDS store set.
#pragma unroll
    for (int nt = 0; nt < RZT; ++nt) {
      v8f zero = {};
      v8f acc = __builtin_amdgcn_wmma_f32_16x16x32_f16(
          false, ax, false, wih_rz[nt], (short)0, zero, false, false);
      acc = __builtin_amdgcn_wmma_f32_16x16x32_f16(
          false, ah, false, whh_rz[nt], (short)0, acc, false, false);
#pragma unroll
      for (int r = 0; r < 8; ++r)
        s_grz[r + kbase][nt * 16 + n_loc] = acc[r];   // lanes 16-31: M = r+8
    }
    // n tiles: keep gx / gh separate (gate needs r * hn).
#pragma unroll
    for (int nt = 0; nt < NTT; ++nt) {
      v8f zero = {};
      v8f dx = __builtin_amdgcn_wmma_f32_16x16x32_f16(
          false, ax, false, wih_n[nt], (short)0, zero, false, false);
      v8f dh = __builtin_amdgcn_wmma_f32_16x16x32_f16(
          false, ah, false, whh_n[nt], (short)0, zero, false, false);
#pragma unroll
      for (int r = 0; r < 8; ++r) {
        s_gnx[r + kbase][nt * 16 + n_loc] = dx[r];
        s_gnh[r + kbase][nt * 16 + n_loc] = dh[r];
      }
    }
    __syncthreads();

    // Gate math: thread slot i handles fixed (b, j).
#pragma unroll
    for (int i = 0; i < GITER; ++i) {
      int idx = i * 32 + lane;
      int b   = idx / HID;
      int j   = idx - b * HID;
      if (b < 16) {
        float r = sigmoidf_(s_grz[b][j] + s_bsum[j]);
        float z = sigmoidf_(s_grz[b][HID + j] + s_bsum[HID + j]);
        float n = tanhf(s_gnx[b][j] + s_bnx[j] + r * (s_gnh[b][j] + s_bnh[j]));
        float h = (1.0f - z) * n + z * hreg[i];
        hreg[i] = h;
        _Float16 h16 = (_Float16)h;
        s_h[b][j] = h16;
        act_out[((long long)(b0 + b) * T_ + t) * 32 + j] = h16;
      }
    }
    __syncthreads();
  }
}

// ---------------------------------------------------------------------------
// FC (16 -> 10) + transpose to [B, NC, T]. Bandwidth-bound; plain VALU.
// ---------------------------------------------------------------------------
__global__ void fc_transpose(const _Float16* __restrict__ act,
                             const float* __restrict__ fw,
                             const float* __restrict__ fb,
                             float* __restrict__ out) {
  long long idx = (long long)blockIdx.x * blockDim.x + threadIdx.x;  // over B*T
  if (idx >= (long long)B_ * T_) return;
  long long b = idx / T_, t = idx - b * T_;
  const _Float16* a = act + idx * 32;
  float av[FCD];
#pragma unroll
  for (int k = 0; k < FCD; ++k) av[k] = (float)a[k];
#pragma unroll
  for (int c = 0; c < NC_; ++c) {
    float s = fb[c];
#pragma unroll
    for (int k = 0; k < FCD; ++k) s += av[k] * fw[c * FCD + k];
    out[(b * NC_ + c) * T_ + t] = s;
  }
}

// ---------------------------------------------------------------------------
// Launch. Input order: x, fc_w, fc_b, then (w_ih, w_hh, b_ih, b_hh) x4.
// ---------------------------------------------------------------------------
extern "C" void kernel_launch(void* const* d_in, const int* in_sizes, int n_in,
                              void* d_out, int out_size, void* d_ws, size_t ws_size,
                              hipStream_t stream) {
  const float* x    = (const float*)d_in[0];
  const float* fc_w = (const float*)d_in[1];
  const float* fc_b = (const float*)d_in[2];
  const float* wih[4] = {(const float*)d_in[3],  (const float*)d_in[7],
                         (const float*)d_in[11], (const float*)d_in[15]};
  const float* whh[4] = {(const float*)d_in[4],  (const float*)d_in[8],
                         (const float*)d_in[12], (const float*)d_in[16]};
  const float* bih[4] = {(const float*)d_in[5],  (const float*)d_in[9],
                         (const float*)d_in[13], (const float*)d_in[17]};
  const float* bhh[4] = {(const float*)d_in[6],  (const float*)d_in[10],
                         (const float*)d_in[14], (const float*)d_in[18]};

  // Workspace layout: two ping-pong activation buffers + packed weights.
  char*  ws       = (char*)d_ws;
  size_t actElems = (size_t)B_ * T_ * 32;
  _Float16* actA = (_Float16*)ws; ws += actElems * sizeof(_Float16);
  _Float16* actB = (_Float16*)ws; ws += actElems * sizeof(_Float16);

  const int HIDs[4] = {30, 15, 30, 16};
  const int INs[4]  = {32, 30, 15, 30};
  const int RZPs[4] = {64, 32, 64, 32};   // ceil(2H/16)*16
  const int NPs[4]  = {32, 16, 32, 16};   // ceil(H/16)*16
  _Float16* wihp[4];
  _Float16* whhp[4];
  for (int l = 0; l < 4; ++l) {
    int GP = RZPs[l] + NPs[l];
    wihp[l] = (_Float16*)ws; ws += (size_t)GP * 32 * sizeof(_Float16);
    whhp[l] = (_Float16*)ws; ws += (size_t)GP * 32 * sizeof(_Float16);
  }

  int n = (int)actElems;
  f32_to_f16<<<(n + 255) / 256, 256, 0, stream>>>(x, actA, n);
  for (int l = 0; l < 4; ++l) {
    int nel = (RZPs[l] + NPs[l]) * 32;
    pack_weights_rzn<<<(nel + 255) / 256, 256, 0, stream>>>(
        wih[l], wihp[l], HIDs[l], INs[l], RZPs[l], NPs[l]);
    pack_weights_rzn<<<(nel + 255) / 256, 256, 0, stream>>>(
        whh[l], whhp[l], HIDs[l], HIDs[l], RZPs[l], NPs[l]);
  }

  gru_scan<30, 4, 2><<<B_ / 16, 32, 0, stream>>>(actA, actB, wihp[0], whhp[0], bih[0], bhh[0]);
  gru_scan<15, 2, 1><<<B_ / 16, 32, 0, stream>>>(actB, actA, wihp[1], whhp[1], bih[1], bhh[1]);
  gru_scan<30, 4, 2><<<B_ / 16, 32, 0, stream>>>(actA, actB, wihp[2], whhp[2], bih[2], bhh[2]);
  gru_scan<16, 2, 1><<<B_ / 16, 32, 0, stream>>>(actB, actA, wihp[3], whhp[3], bih[3], bhh[3]);

  long long bt = (long long)B_ * T_;
  fc_transpose<<<(int)((bt + 255) / 256), 256, 0, stream>>>(actA, fc_w, fc_b, (float*)d_out);
}
